// GIB_large_6794638262418
// MI455X (gfx1250) — compile-verified, matrix operand order checked by reading, not run on hardware
//
#include <hip/hip_runtime.h>
#include <hip/hip_bf16.h>

#define NN   8192
#define DIN  512
#define HH   256
#define DOUT 16
#define KC   16

typedef __attribute__((ext_vector_type(16))) _Float16 v16h;
typedef __attribute__((ext_vector_type(8)))  _Float16 v8h;
typedef __attribute__((ext_vector_type(8)))  float    v8f;
typedef __attribute__((ext_vector_type(4)))  unsigned int v4u;
typedef __attribute__((ext_vector_type(8)))  int      v8i;
typedef __attribute__((ext_vector_type(4)))  int      v4i;

// ---- WMMA fragment loader (ISA 7.12.2 16-bit A/B layout) -------------------
//   idx = lane&15 selects the row (A: M, B: N from an N-major buffer)
//   half = lane>>4: elems 0..7 -> K=half*8+e, elems 8..15 -> K=16+half*8+e
__device__ __forceinline__ v16h load_frag(const _Float16* base, int stride, int lane) {
    const int idx  = lane & 15;
    const int half = (lane >> 4) & 1;
    const _Float16* p = base + (size_t)idx * stride;
    v8h lo = *(const v8h*)(p + half * 8);
    v8h hi = *(const v8h*)(p + 16 + half * 8);
    v16h r;
#pragma unroll
    for (int e = 0; e < 8; ++e) { r[e] = lo[e]; r[8 + e] = hi[e]; }
    return r;
}

__device__ __forceinline__ v8f wmma_f16(v16h a, v16h b, v8f c) {
    return __builtin_amdgcn_wmma_f32_16x16x32_f16(false, a, false, b, (short)0, c,
                                                  false, false);
}

// ---- Tensor Data Mover: 2D f32 tile Global -> LDS (D# per ISA 08 §8.3/8.4) -
// tile_d0 elements per row (contiguous), tile_d1 rows, row stride = stride_elts.
__device__ __forceinline__ void tdm_load_2d_f32(unsigned lds_addr, const void* gptr,
                                                unsigned tile_d0, unsigned tile_d1,
                                                unsigned stride_elts) {
    unsigned long long ga = (unsigned long long)(uintptr_t)gptr;
    v4u g0;
    g0[0] = 1u;                                              // count=1, user desc
    g0[1] = lds_addr;                                        // D#.lds_addr
    g0[2] = (unsigned)ga;                                    // global_addr[31:0]
    g0[3] = (unsigned)((ga >> 32) & 0x1FFFFFFu) | (2u << 30); // [56:32] | type=2
    v8i g1;
    g1[0] = (int)(2u << 16);                                 // data_size=4B
    g1[1] = (int)(((unsigned)NN & 0xffffu) << 16);           // tensor_dim0 lo16
    g1[2] = (int)((((unsigned)NN >> 16) & 0xffffu) |
                  (((unsigned)NN & 0xffffu) << 16));         // d0 hi16 | d1 lo16
    g1[3] = (int)((((unsigned)NN >> 16) & 0xffffu) |
                  (tile_d0 << 16));                          // d1 hi16 | tile_dim0
    g1[4] = (int)(tile_d1 & 0xffffu);                        // tile_dim1 (dim2=0)
    g1[5] = (int)stride_elts;                                // tensor_dim0_stride
    g1[6] = 0;
    g1[7] = 0;
    v4i gz = {0, 0, 0, 0};
#if defined(__clang_major__) && (__clang_major__ >= 23)
    v8i gz8 = {0, 0, 0, 0, 0, 0, 0, 0};
    __builtin_amdgcn_tensor_load_to_lds(g0, g1, gz, gz, gz8, 0);
#else
    __builtin_amdgcn_tensor_load_to_lds(g0, g1, gz, gz, 0);
#endif
}

// ---- K1: Wh = x @ W  -> stored transposed f16 WhT[H][N] --------------------
__global__ void __launch_bounds__(256) k1_gemm_xw(const float* __restrict__ x,
                                                  const float* __restrict__ W,
                                                  _Float16* __restrict__ WhT) {
    __shared__ __align__(16) _Float16 ldsA[128 * 32];   // [m][k]
    __shared__ __align__(16) _Float16 ldsB[128 * 32];   // [n][k] (transposed)
    const int t = threadIdx.x, lane = t & 31, w = t >> 5;
    const int i0 = blockIdx.x * 128, n0 = blockIdx.y * 128;
    const int wm = w >> 1, wn = w & 1;
    const int idx = lane & 15, half = lane >> 4;

    v8f acc[2][4] = {};
    for (int k0 = 0; k0 < DIN; k0 += 32) {
        {
            int r = t >> 1, c = (t & 1) * 16;
            const float* src = x + (size_t)(i0 + r) * DIN + k0 + c;
#pragma unroll
            for (int e = 0; e < 16; ++e) ldsA[r * 32 + c + e] = (_Float16)src[e];
        }
        {
            int k = t >> 3, np = (t & 7) * 16;
            const float* src = W + (size_t)(k0 + k) * HH + n0 + np;
#pragma unroll
            for (int e = 0; e < 16; ++e) ldsB[(np + e) * 32 + k] = (_Float16)src[e];
        }
        __syncthreads();
        v16h af[2], bf[4];
#pragma unroll
        for (int i = 0; i < 2; ++i) af[i] = load_frag(ldsA + (wm * 32 + i * 16) * 32, 32, lane);
#pragma unroll
        for (int j = 0; j < 4; ++j) bf[j] = load_frag(ldsB + (wn * 64 + j * 16) * 32, 32, lane);
#pragma unroll
        for (int i = 0; i < 2; ++i)
#pragma unroll
            for (int j = 0; j < 4; ++j) acc[i][j] = wmma_f16(af[i], bf[j], acc[i][j]);
        __syncthreads();
    }
#pragma unroll
    for (int i = 0; i < 2; ++i)
#pragma unroll
        for (int j = 0; j < 4; ++j)
#pragma unroll
            for (int r = 0; r < 8; ++r) {
                int gi = i0 + wm * 32 + i * 16 + r + half * 8;
                int h  = n0 + wn * 64 + j * 16 + idx;
                WhT[(size_t)h * NN + gi] = (_Float16)acc[i][j][r];
            }
}

// ---- K2: Wh1 = Wh@a[:H], Wh2 = Wh@a[H:] ------------------------------------
__global__ void __launch_bounds__(256) k2_wh_a(const _Float16* __restrict__ WhT,
                                               const float* __restrict__ a,
                                               float* __restrict__ Wh1v,
                                               float* __restrict__ Wh2v) {
    int n = blockIdx.x * 256 + threadIdx.x;
    float s1 = 0.f, s2 = 0.f;
    for (int h = 0; h < HH; ++h) {
        float w = (float)WhT[(size_t)h * NN + n];
        s1 += w * a[h];
        s2 += w * a[HH + h];
    }
    Wh1v[n] = s1;
    Wh2v[n] = s2;
}

// ---- K3: per-row softmax stats over masked e (adj pass 1) ------------------
__global__ void __launch_bounds__(256) k3_stats(const float* __restrict__ adj,
                                                const float* __restrict__ Wh1v,
                                                const float* __restrict__ Wh2v,
                                                float* __restrict__ rowmax,
                                                float* __restrict__ rowinv) {
    __shared__ float redA[8], redB[8];
    const int t = threadIdx.x, lane = t & 31, w = t >> 5;
    const int i = blockIdx.x;
    const float wh1 = Wh1v[i];
    const float* arow = adj + (size_t)i * NN;
    float ev[32];
    float mx = -3.0e38f;
#pragma unroll
    for (int c = 0; c < 32; ++c) {
        int j = c * 256 + t;
        float e = wh1 + Wh2v[j];
        e = e > 0.f ? e : 0.2f * e;
        e = arow[j] > 0.f ? e : -9.0e15f;
        ev[c] = e;
        mx = fmaxf(mx, e);
    }
#pragma unroll
    for (int off = 16; off >= 1; off >>= 1) mx = fmaxf(mx, __shfl_xor(mx, off, 32));
    if (lane == 0) redA[w] = mx;
    __syncthreads();
    mx = redA[0];
#pragma unroll
    for (int k = 1; k < 8; ++k) mx = fmaxf(mx, redA[k]);

    float s = 0.f;
#pragma unroll
    for (int c = 0; c < 32; ++c) s += __expf(ev[c] - mx);
#pragma unroll
    for (int off = 16; off >= 1; off >>= 1) s += __shfl_xor(s, off, 32);
    if (lane == 0) redB[w] = s;
    __syncthreads();
    if (t == 0) {
        float tot = 0.f;
#pragma unroll
        for (int k = 0; k < 8; ++k) tot += redB[k];
        rowmax[i] = mx;
        rowinv[i] = 1.0f / tot;
    }
}

// ---- K4: Z0 = softmax(masked e) @ Wh (adj pass 2) --------------------------
// adj tiles (64x32 f32) streamed by the Tensor Data Mover, double-buffered.
__global__ void __launch_bounds__(256) k4_attn(const float* __restrict__ adj,
                                               const _Float16* __restrict__ WhT,
                                               const float* __restrict__ Wh1v,
                                               const float* __restrict__ Wh2v,
                                               const float* __restrict__ rowmax,
                                               const float* __restrict__ rowinv,
                                               float* __restrict__ Z0) {
    __shared__ __align__(16) _Float16 ldsP[64 * 32];
    __shared__ __align__(16) float ldsAdj[2][64 * 32];
    const int t = threadIdx.x, lane = t & 31, w = t >> 5;
    const int i0 = blockIdx.x * 64;
    const int wm = w >> 1, wn = w & 1;
    const int idx = lane & 15, half = lane >> 4;
    const int pr = t >> 2, pc = (t & 3) * 8;
    const float wh1 = Wh1v[i0 + pr];
    const float mrow = rowmax[i0 + pr];
    const float* tilebase = adj + (size_t)i0 * NN;

    if (w == 0)   // TDM is a per-wave DMA issue; EXEC ignored, wave 0 drives it
        tdm_load_2d_f32((unsigned)(uintptr_t)&ldsAdj[0][0], tilebase, 32, 64, NN);

    v8f acc[8] = {};
    const int nit = NN / 32;
    for (int it = 0; it < nit; ++it) {
        const int j0 = it * 32;
        if (w == 0) {
            if (it + 1 < nit) {
                tdm_load_2d_f32((unsigned)(uintptr_t)&ldsAdj[(it + 1) & 1][0],
                                tilebase + j0 + 32, 32, 64, NN);
                __builtin_amdgcn_s_wait_tensorcnt((short)1);
            } else {
                __builtin_amdgcn_s_wait_tensorcnt((short)0);
            }
        }
        __syncthreads();                       // tile `it` visible to all waves
        const float* at = &ldsAdj[it & 1][0];
#pragma unroll
        for (int e = 0; e < 8; ++e) {
            int j = j0 + pc + e;
            float av = at[pr * 32 + pc + e];
            float ev = wh1 + Wh2v[j];
            ev = ev > 0.f ? ev : 0.2f * ev;
            float p = av > 0.f ? __expf(ev - mrow) : 0.f;
            ldsP[pr * 32 + pc + e] = (_Float16)p;
        }
        __syncthreads();
        v16h af = load_frag(ldsP + (wm * 16) * 32, 32, lane);
#pragma unroll
        for (int jt = 0; jt < 8; ++jt) {
            v16h bf = load_frag(WhT + (size_t)(wn * 128 + jt * 16) * NN + j0, NN, lane);
            acc[jt] = wmma_f16(af, bf, acc[jt]);
        }
        __syncthreads();
    }
#pragma unroll
    for (int jt = 0; jt < 8; ++jt)
#pragma unroll
        for (int r = 0; r < 8; ++r) {
            int gi = i0 + wm * 16 + r + half * 8;
            int h  = wn * 128 + jt * 16 + idx;
            Z0[(size_t)gi * HH + h] = acc[jt][r] * rowinv[gi];
        }
}

// ---- K5: G = elu(Z0) @ W_gcn -> stored transposed f16 GT[16][N] ------------
__global__ void __launch_bounds__(256) k5_gcn_mm(const float* __restrict__ Z0,
                                                 const float* __restrict__ Wgcn,
                                                 _Float16* __restrict__ GT) {
    __shared__ __align__(16) _Float16 ldsW[16 * 256];
    const int t = threadIdx.x, lane = t & 31, w = t >> 5;
    {
        int k = t;
#pragma unroll
        for (int e = 0; e < 16; ++e) ldsW[e * 256 + k] = (_Float16)Wgcn[k * 16 + e];
    }
    __syncthreads();
    const int gi0 = blockIdx.x * 128 + w * 16;
    const int idx = lane & 15, half = lane >> 4;
    v8f acc = {};
    const float* zr = Z0 + (size_t)(gi0 + idx) * HH;
    for (int k0 = 0; k0 < HH; k0 += 32) {
        v16h af;
#pragma unroll
        for (int e = 0; e < 8; ++e) {
            float lo = zr[k0 + half * 8 + e];
            float hi = zr[k0 + 16 + half * 8 + e];
            lo = lo > 0.f ? lo : __expf(lo) - 1.f;
            hi = hi > 0.f ? hi : __expf(hi) - 1.f;
            af[e] = (_Float16)lo;
            af[8 + e] = (_Float16)hi;
        }
        v16h bf = load_frag(ldsW + k0, 256, lane);
        acc = wmma_f16(af, bf, acc);
    }
#pragma unroll
    for (int r = 0; r < 8; ++r) {
        int gi = gi0 + r + half * 8;
        GT[(size_t)idx * NN + gi] = (_Float16)acc[r];
    }
}

// ---- K6: out = Z1 = relu(adj @ G) (adj pass 3) ------------------------------
// adj tiles (128x32 f32) streamed via async global->LDS, double-buffered,
// paced with ASYNCcnt.
__device__ __forceinline__ void k6_stage(const float* __restrict__ base /*uniform*/,
                                         float* ldsbuf, int t) {
#pragma unroll
    for (int q = 0; q < 4; ++q) {
        int fid = q * 1024 + t * 4;                      // 4 consecutive floats
        int r = fid >> 5, c = fid & 31;
        unsigned goff  = (unsigned)((r * NN + c) * 4);   // byte offset, < 2^32
        unsigned laddr = (unsigned)(uintptr_t)(ldsbuf + fid);
        asm volatile("global_load_async_to_lds_b128 %0, %1, %2"
                     :: "v"(laddr), "v"(goff), "s"(base)
                     : "memory");
    }
}

__global__ void __launch_bounds__(256) k6_prop(const float* __restrict__ adj,
                                               const _Float16* __restrict__ GT,
                                               float* __restrict__ out0,
                                               float* __restrict__ out1) {
    __shared__ __align__(16) float ldsAdj[2][128 * 32];
    const int t = threadIdx.x, lane = t & 31, w = t >> 5;
    const int gi0 = blockIdx.x * 128 + w * 16;
    const int idx = lane & 15, half = lane >> 4;
    const float* blockbase = adj + (size_t)(blockIdx.x * 128) * NN;   // uniform

    k6_stage(blockbase, &ldsAdj[0][0], t);               // prologue: tile 0

    v8f acc = {};
    const int nit = NN / 32;
    for (int it = 0; it < nit; ++it) {
        const int j0 = it * 32;
        if (it + 1 < nit) {
            k6_stage(blockbase + j0 + 32, &ldsAdj[(it + 1) & 1][0], t);
            asm volatile("s_wait_asynccnt 0x4" ::: "memory");  // tile `it` done
        } else {
            asm volatile("s_wait_asynccnt 0x0" ::: "memory");  // drain
        }
        __syncthreads();
        const float* af32 = &ldsAdj[it & 1][(w * 16 + idx) * 32];
        v16h af;
#pragma unroll
        for (int e = 0; e < 8; ++e) {                    // adj is exactly 0/1
            af[e]     = (_Float16)af32[half * 8 + e];
            af[8 + e] = (_Float16)af32[16 + half * 8 + e];
        }
        v16h bf = load_frag(GT + j0, NN, lane);          // GT: 256KB, L2-resident
        acc = wmma_f16(af, bf, acc);
        __syncthreads();                                 // before buf overwrite
    }
#pragma unroll
    for (int r = 0; r < 8; ++r) {
        int gi = gi0 + r + half * 8;
        float v = fmaxf(acc[r], 0.f);
        out0[(size_t)gi * DOUT + idx] = v;
        out1[(size_t)gi * DOUT + idx] = v;
    }
}

// ---- K7: cluster score ------------------------------------------------------
__global__ void __launch_bounds__(256) k7_score(const float* __restrict__ Z0,
                                                const float* __restrict__ Ca,
                                                float* __restrict__ score) {
    __shared__ float sc[8][16];
    const int t = threadIdx.x, lane = t & 31, w = t >> 5;
    const int n = blockIdx.x * 8 + w;
    float z[8];
#pragma unroll
    for (int e = 0; e < 8; ++e) z[e] = Z0[(size_t)n * HH + lane * 8 + e];
    float ssum = 0.f;
    for (int k = 0; k < KC; ++k) {
        float d = 0.f;
#pragma unroll
        for (int e = 0; e < 8; ++e) {
            float df = z[e] - Ca[k * HH + lane * 8 + e];
            d += df * df;
        }
#pragma unroll
        for (int off = 16; off >= 1; off >>= 1) d += __shfl_xor(d, off, 32);
        float s = __expf(-sqrtf(d)) + 1e-10f;
        ssum += s;
        if (lane == 0) sc[w][k] = s;
    }
    __syncthreads();
    if (lane < 16) score[(size_t)n * KC + lane] = sc[w][lane] / ssum;
}

extern "C" void kernel_launch(void* const* d_in, const int* in_sizes, int n_in,
                              void* d_out, int out_size, void* d_ws, size_t ws_size,
                              hipStream_t stream) {
    const float* x    = (const float*)d_in[0];   // [8192,512]
    const float* adj  = (const float*)d_in[1];   // [8192,8192]
    const float* Ca   = (const float*)d_in[2];   // [16,256]
    const float* W    = (const float*)d_in[3];   // [512,256]
    const float* a    = (const float*)d_in[4];   // [512,1]
    const float* Wgcn = (const float*)d_in[5];   // [256,16]

    float* out0  = (float*)d_out;                // output  [8192,16]
    float* out1  = out0 + (size_t)NN * DOUT;     // Z_1     [8192,16]
    float* Z0    = out1 + (size_t)NN * DOUT;     // Z_0     [8192,256]
    float* score = Z0 + (size_t)NN * HH;         // score   [8192,16]

    _Float16* WhT = (_Float16*)d_ws;             // [256][8192] f16  (4 MB)
    _Float16* GT  = WhT + (size_t)HH * NN;       // [16][8192]  f16  (256 KB)
    float* Wh1v   = (float*)(GT + (size_t)DOUT * NN);
    float* Wh2v   = Wh1v + NN;
    float* rowmax = Wh2v + NN;
    float* rowinv = rowmax + NN;

    k1_gemm_xw<<<dim3(64, 2), 256, 0, stream>>>(x, W, WhT);
    k2_wh_a<<<32, 256, 0, stream>>>(WhT, a, Wh1v, Wh2v);
    k3_stats<<<NN, 256, 0, stream>>>(adj, Wh1v, Wh2v, rowmax, rowinv);
    k4_attn<<<128, 256, 0, stream>>>(adj, WhT, Wh1v, Wh2v, rowmax, rowinv, Z0);
    k5_gcn_mm<<<64, 256, 0, stream>>>(Z0, Wgcn, GT);
    k6_prop<<<64, 256, 0, stream>>>(adj, GT, out0, out1);
    k7_score<<<1024, 256, 0, stream>>>(Z0, Ca, score);
}
